// SpatialTransformerLayer_50981261803720
// MI455X (gfx1250) — matrix operand order repaired and checked by reading, not compiled, not run on hardware
//
#include <hip/hip_runtime.h>
#include <math.h>

typedef unsigned short u16;
typedef __attribute__((ext_vector_type(16))) __bf16 v16bf;
typedef __attribute__((ext_vector_type(8)))  float  v8f;
typedef __attribute__((ext_vector_type(4)))  int    v4i;

union FragBF { uint4 q[2]; v16bf v; };
union V16U   { uint4 q[2]; u16 s[16]; };

__device__ __forceinline__ u16 f2bf(float f) {
    union { float f; unsigned u; } c; c.f = f;
    unsigned u = c.u;
    u += 0x7FFFu + ((u >> 16) & 1u);     // round-to-nearest-even
    return (u16)(u >> 16);
}

__device__ __forceinline__ v8f vzero() {
    v8f z;
#pragma unroll
    for (int i = 0; i < 8; ++i) z[i] = 0.f;
    return z;
}

// Async DMA: global -> LDS, b128 per lane, tracked by ASYNCcnt (no VGPR staging)
__device__ __forceinline__ void async_ld_b128(const u16* g, u16* l) {
    __builtin_amdgcn_global_load_async_to_lds_b128(
        (__attribute__((address_space(1))) v4i*)g,
        (__attribute__((address_space(3))) v4i*)l, 0, 0);
}
__device__ __forceinline__ void wait_async0() {
    asm volatile("s_wait_asynccnt 0x0" ::: "memory");
}

// Load one 16-element bf16 fragment as two b128 LDS reads (chunks may be disjoint)
__device__ __forceinline__ v16bf load_frag2(const u16* p0, const u16* p1) {
    FragBF f;
    f.q[0] = *reinterpret_cast<const uint4*>(p0);
    f.q[1] = *reinterpret_cast<const uint4*>(p1);
    return f.v;
}

__device__ __forceinline__ float gelu_exact(float v) {
    return 0.5f * v * (1.f + erff(v * 0.70710678118654752f));
}

// ---------------------------------------------------------------- convert f32 -> bf16
__global__ __launch_bounds__(256) void convert_bf16_kernel(
    const float* __restrict__ src, u16* __restrict__ dst, int n) {
    int i = blockIdx.x * 256 + threadIdx.x;
    int stride = gridDim.x * 256;
    for (; i < n; i += stride) dst[i] = f2bf(src[i]);
}

// ---------------------------------------------------------------- layernorm, 768 wide
__global__ __launch_bounds__(256) void layernorm_bf16_kernel(
    const float* __restrict__ x, const float* __restrict__ w,
    const float* __restrict__ b, u16* __restrict__ out) {
    const int row = blockIdx.x;
    const int tid = threadIdx.x;
    const float* xr = x + (size_t)row * 768;
    float v0 = xr[tid], v1 = xr[tid + 256], v2 = xr[tid + 512];
    float s  = v0 + v1 + v2;
    float sq = v0 * v0 + v1 * v1 + v2 * v2;
#pragma unroll
    for (int off = 16; off > 0; off >>= 1) {
        s  += __shfl_down(s, off, 32);
        sq += __shfl_down(sq, off, 32);
    }
    __shared__ float ps[8], pq[8];
    __shared__ float smu, srs;
    int wv = tid >> 5, ln = tid & 31;
    if (ln == 0) { ps[wv] = s; pq[wv] = sq; }
    __syncthreads();
    if (tid == 0) {
        float ts = 0.f, tq = 0.f;
        for (int i = 0; i < 8; ++i) { ts += ps[i]; tq += pq[i]; }
        float mu = ts * (1.f / 768.f);
        float var = tq * (1.f / 768.f) - mu * mu;
        smu = mu; srs = rsqrtf(var + 1e-5f);
    }
    __syncthreads();
    float mu = smu, rs = srs;
    u16* orow = out + (size_t)row * 768;
    orow[tid]       = f2bf((v0 - mu) * rs * w[tid]       + b[tid]);
    orow[tid + 256] = f2bf((v1 - mu) * rs * w[tid + 256] + b[tid + 256]);
    orow[tid + 512] = f2bf((v2 - mu) * rs * w[tid + 512] + b[tid + 512]);
}

// ---------------------------------------------------------------- WMMA GEMM
// C[M,N] = A[M,K] * W[N,K]^T (+bias)(+act)(+residual), A/W bf16, fp32 accum.
// Tiles stream global->LDS via GLOBAL_LOAD_ASYNC_TO_LDS_B128 (double buffered).
#define BM 128
#define BN 128
#define BK 32
#define LDT 40   // LDS row stride (u16 elems): 80B rows keep b128 chunks aligned

template<int ACT, bool BIAS, bool RES, bool OUTBF>
__global__ __launch_bounds__(256) void gemm_wmma_kernel(
    const u16* __restrict__ A, const u16* __restrict__ W,
    const float* __restrict__ bias, const float* __restrict__ res,
    u16* __restrict__ outb, float* __restrict__ outf,
    int M, int N, int K) {
    __shared__ u16 As[2][BM * LDT];
    __shared__ u16 Bs[2][BN * LDT];

    const int tid  = threadIdx.x;
    const int lane = tid & 31;
    const int w    = tid >> 5;
    const int wrow = w >> 1;          // 0..3  (32-row slab)
    const int wcol = w & 1;           // 0..1  (64-col slab)
    const int bm = blockIdx.y * BM;
    const int bn = blockIdx.x * BN;

    // cooperative tile DMA: each thread owns one row-half (16 bf16 = 2x b128)
    const int lrow = tid >> 1;
    const int lch  = (tid & 1) * 16;
    const size_t aoff = (size_t)(bm + lrow) * K + lch;
    const size_t boff = (size_t)(bn + lrow) * K + lch;
    u16* lA = &As[0][lrow * LDT + lch];
    u16* lB = &Bs[0][lrow * LDT + lch];

    v8f acc[2][4];
#pragma unroll
    for (int i = 0; i < 2; ++i)
#pragma unroll
        for (int j = 0; j < 4; ++j) acc[i][j] = vzero();

    const int KT = K / BK;
    // issue tile kt into buffer b (4 async b128 per thread, no VGPR staging)
    auto issue_tile = [&](int kt, int b) {
        const u16* ga = A + aoff + (size_t)kt * BK;
        const u16* gb = W + boff + (size_t)kt * BK;
        u16* la = lA + b * (BM * LDT);
        u16* lb = lB + b * (BN * LDT);
        async_ld_b128(ga,     la);
        async_ld_b128(ga + 8, la + 8);
        async_ld_b128(gb,     lb);
        async_ld_b128(gb + 8, lb + 8);
    };
    issue_tile(0, 0);

    const int c0 = (lane >> 4) * 8;    // A-frag K-chunk base (ISA 16-bit A layout)
    const int kb = (lane >> 4) * 16;   // unused; B frags use c0 (same k order as A)
    int buf = 0;
    for (int kt = 0; kt < KT; ++kt) {
        wait_async0();                 // this wave's DMA for tile kt complete
        __syncthreads();               // all waves' DMA complete + prev reads done
        if (kt + 1 < KT) {
            issue_tile(kt + 1, buf ^ 1);   // DMA next tile while we compute
            if (kt + 2 < KT) {
                __builtin_prefetch(A + aoff + (size_t)(kt + 2) * BK, 0, 1);
                __builtin_prefetch(W + boff + (size_t)(kt + 2) * BK, 0, 1);
            }
        }
        v16bf af[2], bfr[4];
#pragma unroll
        for (int mi = 0; mi < 2; ++mi) {
            int r = (wrow * 2 + mi) * 16 + (lane & 15);
            af[mi] = load_frag2(&As[buf][r * LDT + c0], &As[buf][r * LDT + c0 + 16]);
        }
#pragma unroll
        for (int nj = 0; nj < 4; ++nj) {
            int c = (wcol * 4 + nj) * 16 + (lane & 15);
            bfr[nj] = load_frag2(&Bs[buf][c * LDT + c0], &Bs[buf][c * LDT + c0 + 16]);
        }
#pragma unroll
        for (int mi = 0; mi < 2; ++mi)
#pragma unroll
            for (int nj = 0; nj < 4; ++nj)
                acc[mi][nj] = __builtin_amdgcn_wmma_f32_16x16x32_bf16(
                    false, af[mi], false, bfr[nj], (short)0, acc[mi][nj], false, false);
        buf ^= 1;
    }

    // epilogue: C layout — VGPR j <-> row j + 8*(lane>=16), col = lane&15
    const int subrow = (lane >> 4) << 3;
    const int col0   = lane & 15;
#pragma unroll
    for (int mi = 0; mi < 2; ++mi) {
#pragma unroll
        for (int nj = 0; nj < 4; ++nj) {
            int gcol = bn + (wcol * 4 + nj) * 16 + col0;
            float bv = BIAS ? bias[gcol] : 0.f;
#pragma unroll
            for (int j = 0; j < 8; ++j) {
                int grow = bm + (wrow * 2 + mi) * 16 + subrow + j;
                float v = acc[mi][nj][j];
                if (BIAS) v += bv;
                if (ACT == 1) v = gelu_exact(v);
                if (RES) v += res[(size_t)grow * N + gcol];
                if (OUTBF) outb[(size_t)grow * N + gcol] = f2bf(v);
                else       outf[(size_t)grow * N + gcol] = v;
            }
        }
    }
}

// ---------------------------------------------------------------- flash attention
// qkv: [M=8192][2304] bf16 (q | k | v per head). 64 queries/block, 64-key tiles.
// Q/K tiles stream async to LDS; V is transposed through VGPRs into LDS.
#define ALD 72   // LDS row stride (144B rows, b128-aligned chunks)

__global__ __launch_bounds__(256) void attention_wmma_kernel(
    const u16* __restrict__ qkv, u16* __restrict__ attnout) {
    __shared__ u16  Qs[64 * ALD];
    __shared__ u16  Ks[64 * ALD];
    __shared__ u16  Vt[64 * ALD];   // transposed: [chan][key]
    __shared__ float Ss[64 * ALD];
    __shared__ u16  Ps[64 * ALD];
    __shared__ float alpha_s[64];
    __shared__ float linv_s[64];

    const int tid  = threadIdx.x;
    const int lane = tid & 31;
    const int w    = tid >> 5;
    const int b    = blockIdx.y / 12;
    const int head = blockIdx.y % 12;
    const int tokbase = b * 2048 + blockIdx.x * 64;

    const int c0 = (lane >> 4) * 8;
    const int kb = (lane >> 4) * 16;
    const int subrow = (lane >> 4) << 3;
    const int col0 = lane & 15;
    const float scale = 0.125f;     // 64^-0.5, exact pow2

    const int lr = tid >> 2;            // cooperative tile row (0..63)
    const int lc = (tid & 3) * 16;      // 16-elem chunk within row

    // async DMA of Q tile (64 x 64); completion folded into first loop wait
    {
        const u16* gq = qkv + (size_t)(tokbase + lr) * 2304 + head * 64 + lc;
        u16* lq = &Qs[lr * ALD + lc];
        async_ld_b128(gq,     lq);
        async_ld_b128(gq + 8, lq + 8);
    }

    v8f oacc[2];
    oacc[0] = vzero(); oacc[1] = vzero();
    float ms = -INFINITY, ls = 0.f;

    const int srb = w >> 1;                 // S-phase row block
    const int scb = (w & 1) * 2;            // S-phase col block base (2 tiles)
    const int orb = w & 3;                  // O-phase row block
    const int ocb = (w >> 2) * 2;           // O-phase col block base (2 tiles)

    for (int kt = 0; kt < 32; ++kt) {
        __syncthreads();   // previous iteration's reads of Ks/Vt/Ps complete
        {   // K tile: async DMA to LDS; V tile: load + transpose into Vt
            size_t base = (size_t)(b * 2048 + kt * 64 + lr) * 2304 + head * 64 + lc;
            const u16* gk = qkv + base + 768;
            u16* lk = &Ks[lr * ALD + lc];
            async_ld_b128(gk,     lk);
            async_ld_b128(gk + 8, lk + 8);
            V16U vv;
            vv.q[0] = *(const uint4*)(qkv + base + 1536);
            vv.q[1] = *(const uint4*)(qkv + base + 1536 + 8);
#pragma unroll
            for (int i = 0; i < 16; ++i) Vt[(lc + i) * ALD + lr] = vv.s[i];
        }
        wait_async0();     // Q (iter 0) and K DMA for this wave complete
        __syncthreads();

        // S = Q @ K^T  (per wave: 1 row-block x 2 col-tiles, K-dim 64 = 2 steps)
        v8f sacc[2];
        sacc[0] = vzero(); sacc[1] = vzero();
#pragma unroll
        for (int kk = 0; kk < 64; kk += 32) {
            int qr = srb * 16 + col0;
            v16bf aq = load_frag2(&Qs[qr * ALD + kk + c0], &Qs[qr * ALD + kk + c0 + 16]);
#pragma unroll
            for (int t = 0; t < 2; ++t) {
                int kr = (scb + t) * 16 + col0;
                v16bf bk = load_frag2(&Ks[kr * ALD + kk + c0], &Ks[kr * ALD + kk + c0 + 16]);
                sacc[t] = __builtin_amdgcn_wmma_f32_16x16x32_bf16(
                    false, aq, false, bk, (short)0, sacc[t], false, false);
            }
        }
#pragma unroll
        for (int t = 0; t < 2; ++t)
#pragma unroll
            for (int j = 0; j < 8; ++j)
                Ss[(srb * 16 + subrow + j) * ALD + (scb + t) * 16 + col0] = sacc[t][j];
        __syncthreads();

        // online softmax: one query row per thread (tid < 64)
        if (tid < 64) {
            float tm = -INFINITY;
            for (int j = 0; j < 64; ++j) tm = fmaxf(tm, Ss[tid * ALD + j] * scale);
            float mn = fmaxf(ms, tm);
            float al = __expf(ms - mn);
            float psum = 0.f;
            for (int j = 0; j < 64; ++j) {
                float p = __expf(Ss[tid * ALD + j] * scale - mn);
                Ps[tid * ALD + j] = f2bf(p);
                psum += p;
            }
            ls = ls * al + psum;
            ms = mn;
            alpha_s[tid] = al;
        }
        __syncthreads();

        // rescale running O, then O += P @ V
#pragma unroll
        for (int t = 0; t < 2; ++t)
#pragma unroll
            for (int j = 0; j < 8; ++j)
                oacc[t][j] *= alpha_s[orb * 16 + subrow + j];
#pragma unroll
        for (int kk = 0; kk < 64; kk += 32) {
            int pr = orb * 16 + col0;
            v16bf ap = load_frag2(&Ps[pr * ALD + kk + c0], &Ps[pr * ALD + kk + c0 + 16]);
#pragma unroll
            for (int t = 0; t < 2; ++t) {
                int vc = (ocb + t) * 16 + col0;
                v16bf bv = load_frag2(&Vt[vc * ALD + kk + c0], &Vt[vc * ALD + kk + c0 + 16]);
                oacc[t] = __builtin_amdgcn_wmma_f32_16x16x32_bf16(
                    false, ap, false, bv, (short)0, oacc[t], false, false);
            }
        }
    }

    if (tid < 64) linv_s[tid] = 1.f / ls;
    __syncthreads();
#pragma unroll
    for (int t = 0; t < 2; ++t)
#pragma unroll
        for (int j = 0; j < 8; ++j) {
            int r = orb * 16 + subrow + j;
            int c = (ocb + t) * 16 + col0;
            attnout[(size_t)(tokbase + r) * 768 + head * 64 + c] =
                f2bf(oacc[t][j] * linv_s[r]);
        }
}

// ---------------------------------------------------------------- launch
extern "C" void kernel_launch(void* const* d_in, const int* in_sizes, int n_in,
                              void* d_out, int out_size, void* d_ws, size_t ws_size,
                              hipStream_t stream) {
    const float* x      = (const float*)d_in[0];
    const float* qkv_w  = (const float*)d_in[1];
    const float* proj_w = (const float*)d_in[2];
    const float* proj_b = (const float*)d_in[3];
    const float* n1_w   = (const float*)d_in[4];
    const float* n1_b   = (const float*)d_in[5];
    const float* n2_w   = (const float*)d_in[6];
    const float* n2_b   = (const float*)d_in[7];
    const float* fc1_w  = (const float*)d_in[8];
    const float* fc1_b  = (const float*)d_in[9];
    const float* fc2_w  = (const float*)d_in[10];
    const float* fc2_b  = (const float*)d_in[11];
    float* out = (float*)d_out;

    const int M = 8192;               // B*N tokens
    char* ws = (char*)d_ws;
    size_t off = 0;
    auto alloc = [&](size_t bytes) { char* p = ws + off; off = (off + bytes + 255) & ~(size_t)255; return p; };
    u16* wq  = (u16*)alloc((size_t)2304 * 768 * 2);
    u16* wp  = (u16*)alloc((size_t)768 * 768 * 2);
    u16* w1  = (u16*)alloc((size_t)3072 * 768 * 2);
    u16* w2  = (u16*)alloc((size_t)768 * 3072 * 2);
    u16* hb  = (u16*)alloc((size_t)M * 768 * 2);
    u16* qkvb = (u16*)alloc((size_t)M * 2304 * 2);
    u16* aob = (u16*)alloc((size_t)M * 768 * 2);
    u16* f1b = (u16*)alloc((size_t)M * 3072 * 2);

    // 1) weights -> bf16
    convert_bf16_kernel<<<1024, 256, 0, stream>>>(qkv_w, wq, 2304 * 768);
    convert_bf16_kernel<<<1024, 256, 0, stream>>>(proj_w, wp, 768 * 768);
    convert_bf16_kernel<<<1024, 256, 0, stream>>>(fc1_w, w1, 3072 * 768);
    convert_bf16_kernel<<<1024, 256, 0, stream>>>(fc2_w, w2, 768 * 3072);

    // 2) h = LN1(x) -> bf16
    layernorm_bf16_kernel<<<M, 256, 0, stream>>>(x, n1_w, n1_b, hb);

    // 3) qkv = h @ qkv_w^T
    gemm_wmma_kernel<0, false, false, true><<<dim3(2304 / BN, M / BM), 256, 0, stream>>>(
        hb, wq, nullptr, nullptr, qkvb, nullptr, M, 2304, 768);

    // 4) attention
    attention_wmma_kernel<<<dim3(32, 48), 256, 0, stream>>>(qkvb, aob);

    // 5) x1 = x + attn @ proj_w^T + proj_b   (fp32 into d_out)
    gemm_wmma_kernel<0, true, true, false><<<dim3(768 / BN, M / BM), 256, 0, stream>>>(
        aob, wp, proj_b, x, nullptr, out, M, 768, 768);

    // 6) h2 = LN2(x1) -> bf16 (reuse hb)
    layernorm_bf16_kernel<<<M, 256, 0, stream>>>(out, n2_w, n2_b, hb);

    // 7) f1 = gelu(h2 @ fc1_w^T + fc1_b) -> bf16
    gemm_wmma_kernel<1, true, false, true><<<dim3(3072 / BN, M / BM), 256, 0, stream>>>(
        hb, w1, fc1_b, nullptr, f1b, nullptr, M, 3072, 768);

    // 8) out = x1 + f1 @ fc2_w^T + fc2_b   (reads+writes d_out in place)
    gemm_wmma_kernel<0, true, true, false><<<dim3(768 / BN, M / BM), 256, 0, stream>>>(
        f1b, w2, fc2_b, out, nullptr, out, M, 768, 3072);
}
